// SemSim_3470333575875
// MI455X (gfx1250) — matrix operand order repaired and checked by reading, not compiled, not run on hardware
//
#include <hip/hip_runtime.h>
#include <math.h>

// ---------------------------------------------------------------------------
// Cosine similarity table: out[r,v] = <x_r, e_v> / (max(|x_r|,eps)*max(|e_v|,eps))
// x: [1, 2048, 512] f32   emb: [100000, 512] f32   out: [1, 2048, 100000] f32
//
// GEMM on v_wmma_f32_16x16x32_bf16 with split-bf16 (hi/lo) inputs:
//   a*b ~= ah*bh + ah*bl + al*bh   (f32 accumulate, ~fp32 accuracy)
//
// Grid: M-tiles on blockIdx.x (fast) so the 32 blocks sharing one emb strip
// run temporally adjacent -> emb read from HBM ~once (L2-resident reuse).
// Output stored non-temporally to keep L2 for emb/x.
// 2-stage pipeline: prefetch next chunk to regs while WMMAs run.
// ---------------------------------------------------------------------------

typedef __attribute__((ext_vector_type(4)))  float   v4f;
typedef __attribute__((ext_vector_type(8)))  float   v8f;
typedef __attribute__((ext_vector_type(4)))  __bf16  v4bf;
typedef __attribute__((ext_vector_type(8)))  __bf16  v8bf;
typedef __attribute__((ext_vector_type(16))) __bf16  v16bf;

union FragBF {
    v16bf v;
    v8bf  h[2];   // h[0] -> VGPR0..3, h[1] -> VGPR4..7
};

#define EPSN 1e-8f

static constexpr int R_ROWS = 2048;
static constexpr int V_ROWS = 100000;
static constexpr int DDIM   = 512;

static constexpr int BM     = 64;           // block tile M
static constexpr int BN     = 128;          // block tile N
static constexpr int BK     = 32;           // K chunk (one bf16 WMMA K)
static constexpr int LDA    = 40;           // padded LDS stride (80B, 16B aligned)
static constexpr int NCHUNK = DDIM / BK;    // 16

// ---------------------------------------------------------------------------
// inv-norm: one wave (32 lanes) per row, 8 rows per 256-thread block
// ---------------------------------------------------------------------------
__global__ __launch_bounds__(256)
void rownorm_inv_kernel(const float* __restrict__ src, float* __restrict__ dst,
                        int rows) {
    const int wid  = threadIdx.x >> 5;
    const int lane = threadIdx.x & 31;
    const int row  = blockIdx.x * 8 + wid;
    if (row >= rows) return;

    const float* p = src + (size_t)row * DDIM;
    float s = 0.0f;
#pragma unroll 4
    for (int k = lane; k < DDIM; k += 32) {
        const float v = p[k];
        s = fmaf(v, v, s);
    }
#pragma unroll
    for (int off = 16; off > 0; off >>= 1)
        s += __shfl_xor(s, off, 32);

    if (lane == 0) dst[row] = 1.0f / fmaxf(sqrtf(s), EPSN);
}

// ---------------------------------------------------------------------------
// Tiled WMMA GEMM, split-bf16 inputs, fused cosine scaling, double-buffered LDS
// 256 threads = 8 waves, wave grid 2(M) x 4(N), each wave -> 32x32 outputs
// ---------------------------------------------------------------------------
__global__ __launch_bounds__(256)
void cosine_wmma_kernel(const float* __restrict__ X,     // [2048, 512]
                        const float* __restrict__ E,     // [100000, 512]
                        const float* __restrict__ xinv,  // [2048]
                        const float* __restrict__ einv,  // [100000]
                        float* __restrict__ out) {       // [2048, 100000]
    __shared__ __bf16 As_hi[2][BM * LDA];
    __shared__ __bf16 As_lo[2][BM * LDA];
    __shared__ __bf16 Bs_hi[2][BN * LDA];
    __shared__ __bf16 Bs_lo[2][BN * LDA];

    const int m_block = blockIdx.x * BM;   // fast grid dim: M (L2 reuse of emb)
    const int n_block = blockIdx.y * BN;

    const int tid  = threadIdx.x;
    const int wid  = tid >> 5;
    const int lane = tid & 31;
    const int l16  = lane & 15;
    const int hi16 = lane >> 4;

    const int wm = (wid & 1) * 32;   // wave M origin inside block tile
    const int wn = (wid >> 1) * 32;  // wave N origin inside block tile

    // per-thread staging slots (constant across chunks)
    int a_row[2], a_kq[2];
#pragma unroll
    for (int i = 0; i < 2; ++i) {
        const int q = tid + 256 * i;          // 0..511 : 64 rows x 8 float4
        a_row[i] = q >> 3;
        a_kq[i]  = (q & 7) << 2;
    }
    int b_row[4], b_kq[4];
    bool b_ok[4];
#pragma unroll
    for (int j = 0; j < 4; ++j) {
        const int q = tid + 256 * j;          // 0..1023 : 128 rows x 8 float4
        b_row[j] = q >> 3;
        b_kq[j]  = (q & 7) << 2;
        b_ok[j]  = (n_block + b_row[j]) < V_ROWS;
    }

    v4f aReg[2], bReg[4];

    // ---- prefetch one K-chunk of f32 tiles into registers
    auto load_regs = [&](int kc) {
#pragma unroll
        for (int i = 0; i < 2; ++i)
            aReg[i] = *(const v4f*)&X[(size_t)(m_block + a_row[i]) * DDIM + kc + a_kq[i]];
#pragma unroll
        for (int j = 0; j < 4; ++j) {
            v4f b = { 0.0f, 0.0f, 0.0f, 0.0f };
            if (b_ok[j])
                b = *(const v4f*)&E[(size_t)(n_block + b_row[j]) * DDIM + kc + b_kq[j]];
            bReg[j] = b;
        }
    };

    // ---- split f32 regs -> hi/lo bf16 in LDS buffer `buf`
    auto store_lds = [&](int buf) {
#pragma unroll
        for (int i = 0; i < 2; ++i) {
            const v4f a = aReg[i];
            const v4bf h = { (__bf16)a.x, (__bf16)a.y, (__bf16)a.z, (__bf16)a.w };
            const v4bf l = { (__bf16)(a.x - (float)h.x), (__bf16)(a.y - (float)h.y),
                             (__bf16)(a.z - (float)h.z), (__bf16)(a.w - (float)h.w) };
            *(v4bf*)&As_hi[buf][a_row[i] * LDA + a_kq[i]] = h;
            *(v4bf*)&As_lo[buf][a_row[i] * LDA + a_kq[i]] = l;
        }
#pragma unroll
        for (int j = 0; j < 4; ++j) {
            const v4f b = bReg[j];
            const v4bf h = { (__bf16)b.x, (__bf16)b.y, (__bf16)b.z, (__bf16)b.w };
            const v4bf l = { (__bf16)(b.x - (float)h.x), (__bf16)(b.y - (float)h.y),
                             (__bf16)(b.z - (float)h.z), (__bf16)(b.w - (float)h.w) };
            *(v4bf*)&Bs_hi[buf][b_row[j] * LDA + b_kq[j]] = h;
            *(v4bf*)&Bs_lo[buf][b_row[j] * LDA + b_kq[j]] = l;
        }
    };

    v8f acc[2][2] = {};  // [m subtile][n subtile]

    // ---- pipeline prologue
    load_regs(0);
    store_lds(0);
    __syncthreads();

    for (int ck = 0; ck < NCHUNK; ++ck) {
        const int cur = ck & 1;
        const int nxt = cur ^ 1;

        if (ck + 1 < NCHUNK) load_regs((ck + 1) * BK);   // global loads in flight

        // ---- A fragments (16-bit A layout: lane<16 K={0..7,16..23}, lane>=16 +8)
        FragBF ah[2], al[2];
#pragma unroll
        for (int ms = 0; ms < 2; ++ms) {
            const __bf16* ph = &As_hi[cur][(wm + ms * 16 + l16) * LDA];
            const __bf16* pl = &As_lo[cur][(wm + ms * 16 + l16) * LDA];
            ah[ms].h[0] = *(const v8bf*)(ph + 8 * hi16);
            ah[ms].h[1] = *(const v8bf*)(ph + 16 + 8 * hi16);
            al[ms].h[0] = *(const v8bf*)(pl + 8 * hi16);
            al[ms].h[1] = *(const v8bf*)(pl + 16 + 8 * hi16);
        }

        // ---- B fragments (lane<16: K={0..7,16..23}, lane>=16 +8; N = lane&15)
#pragma unroll
        for (int ns = 0; ns < 2; ++ns) {
            FragBF bh, bl;
            const __bf16* ph = &Bs_hi[cur][(wn + ns * 16 + l16) * LDA + 8 * hi16];
            const __bf16* pl = &Bs_lo[cur][(wn + ns * 16 + l16) * LDA + 8 * hi16];
            bh.h[0] = *(const v8bf*)(ph);
            bh.h[1] = *(const v8bf*)(ph + 16);
            bl.h[0] = *(const v8bf*)(pl);
            bl.h[1] = *(const v8bf*)(pl + 16);

#pragma unroll
            for (int ms = 0; ms < 2; ++ms) {
                acc[ms][ns] = __builtin_amdgcn_wmma_f32_16x16x32_bf16(
                    false, ah[ms].v, false, bh.v, (short)0, acc[ms][ns], false, false);
                acc[ms][ns] = __builtin_amdgcn_wmma_f32_16x16x32_bf16(
                    false, ah[ms].v, false, bl.v, (short)0, acc[ms][ns], false, false);
                acc[ms][ns] = __builtin_amdgcn_wmma_f32_16x16x32_bf16(
                    false, al[ms].v, false, bh.v, (short)0, acc[ms][ns], false, false);
            }
        }

        if (ck + 1 < NCHUNK) store_lds(nxt);             // fill other buffer
        __syncthreads();                                  // one barrier per chunk
    }

    // ---- fused cosine scaling + non-temporal store
    // C/D layout: VGPR i -> M = i (lanes 0-15) / i+8 (lanes 16-31); N = lane&15
#pragma unroll
    for (int ms = 0; ms < 2; ++ms) {
        const int rbase = m_block + wm + ms * 16 + 8 * hi16;
        float xs[8];
#pragma unroll
        for (int i = 0; i < 8; ++i) xs[i] = xinv[rbase + i];

#pragma unroll
        for (int ns = 0; ns < 2; ++ns) {
            const int col = n_block + wn + ns * 16 + l16;
            if (col < V_ROWS) {
                const float es = einv[col];
#pragma unroll
                for (int i = 0; i < 8; ++i) {
                    __builtin_nontemporal_store(acc[ms][ns][i] * xs[i] * es,
                                                &out[(size_t)(rbase + i) * V_ROWS + col]);
                }
            }
        }
    }
}

// ---------------------------------------------------------------------------
extern "C" void kernel_launch(void* const* d_in, const int* in_sizes, int n_in,
                              void* d_out, int out_size, void* d_ws, size_t ws_size,
                              hipStream_t stream) {
    (void)in_sizes; (void)n_in; (void)out_size; (void)ws_size;

    const float* x   = (const float*)d_in[0];  // [1, 2048, 512]
    const float* emb = (const float*)d_in[1];  // [100000, 512]
    float* out  = (float*)d_out;               // [1, 2048, 100000]
    float* xinv = (float*)d_ws;                // [2048]
    float* einv = xinv + R_ROWS;               // [100000]

    // 1) inverse norms (one wave per row)
    rownorm_inv_kernel<<<(R_ROWS + 7) / 8, 256, 0, stream>>>(x, xinv, R_ROWS);
    rownorm_inv_kernel<<<(V_ROWS + 7) / 8, 256, 0, stream>>>(emb, einv, V_ROWS);

    // 2) WMMA GEMM with fused scaling; M-tiles fast for emb L2 reuse
    dim3 grid(R_ROWS / BM, (V_ROWS + BN - 1) / BN);
    cosine_wmma_kernel<<<grid, 256, 0, stream>>>(x, emb, xinv, einv, out);
}
